// CachedMultiheadAttention_65575560675489
// MI455X (gfx1250) — matrix operand order repaired and checked by reading, not compiled, not run on hardware
//
#include <hip/hip_runtime.h>
#include <hip/hip_bf16.h>
#include <stdint.h>

// ---- problem constants (match setup_inputs) ----
#define B_NUM  2
#define T_SEQ  2048
#define E_DIM  1024
#define H_NUM  16
#define HD     64
#define WINDOW 512

typedef __bf16 bf16;
typedef __attribute__((ext_vector_type(16))) __bf16 v16bf;
typedef __attribute__((ext_vector_type(8)))  float  v8f;
typedef unsigned int u32x4 __attribute__((ext_vector_type(4)));
typedef float        f32x4 __attribute__((ext_vector_type(4)));
typedef bf16         bf16x4 __attribute__((ext_vector_type(4)));

// 32-byte WMMA operand fragment (16 bf16 values per lane)
union Frag {
    v16bf v;
    bf16  e[16];
    u32x4 q[2];
};

__device__ inline v8f wmma_bf16(const Frag& a, const Frag& b, v8f c) {
    return __builtin_amdgcn_wmma_f32_16x16x32_bf16(
        /*neg_a=*/false, a.v, /*neg_b=*/false, b.v,
        /*c_mod=*/(short)0, c, /*reuse_a=*/false, /*reuse_b=*/false);
}

// ============================================================================
// Kernel 0: elementwise f32 -> bf16 (one pass; payload fits in L2 afterwards)
// ============================================================================
__global__ __launch_bounds__(256) void cvt_bf16_kernel(
    const float* __restrict__ src, bf16* __restrict__ dst, int n4) {
    int idx    = blockIdx.x * blockDim.x + threadIdx.x;
    int stride = gridDim.x * blockDim.x;
    for (int i = idx; i < n4; i += stride) {
        f32x4 v = ((const f32x4*)src)[i];
        bf16x4 o;
        o.x = (bf16)v.x; o.y = (bf16)v.y; o.z = (bf16)v.z; o.w = (bf16)v.w;
        ((bf16x4*)dst)[i] = o;
    }
}

// ============================================================================
// Shared GEMM machinery: C[M,N] = A[M,K] * W[N,K]^T, all bf16 operands.
// Wave tile: 32(M) x 64(N) = 8 accumulators; k-step 32; register-double-
// buffered so 12 b128 loads for step k+1 are in flight during 8 WMMAs of k.
// ============================================================================
struct TileRegs {
    Frag a[2];   // two 16x32 A fragments (M sub-tiles)
    Frag b[4];   // four 32x16 B fragments (N sub-tiles)
};

// aptr_i = &A[(m0+16i+col)*K] + 16*hi ; bptr_t = &W[(n0+16t+col)*K] + 16*hi
__device__ inline void load_tile(TileRegs& T,
                                 const bf16* __restrict__ a0,
                                 const bf16* __restrict__ a1,
                                 const bf16* __restrict__ b0,
                                 const bf16* __restrict__ b1,
                                 const bf16* __restrict__ b2,
                                 const bf16* __restrict__ b3, int kk) {
    T.a[0].q[0] = *(const u32x4*)(a0 + kk);
    T.a[0].q[1] = *(const u32x4*)(a0 + kk + 8);
    T.a[1].q[0] = *(const u32x4*)(a1 + kk);
    T.a[1].q[1] = *(const u32x4*)(a1 + kk + 8);
    T.b[0].q[0] = *(const u32x4*)(b0 + kk);
    T.b[0].q[1] = *(const u32x4*)(b0 + kk + 8);
    T.b[1].q[0] = *(const u32x4*)(b1 + kk);
    T.b[1].q[1] = *(const u32x4*)(b1 + kk + 8);
    T.b[2].q[0] = *(const u32x4*)(b2 + kk);
    T.b[2].q[1] = *(const u32x4*)(b2 + kk + 8);
    T.b[3].q[0] = *(const u32x4*)(b3 + kk);
    T.b[3].q[1] = *(const u32x4*)(b3 + kk + 8);
}

__device__ inline void mma_tile(v8f (&acc)[2][4], const TileRegs& T) {
#pragma unroll
    for (int i = 0; i < 2; ++i)
#pragma unroll
        for (int t = 0; t < 4; ++t)
            acc[i][t] = wmma_bf16(T.a[i], T.b[t], acc[i][t]);
}

// full K loop with ping-pong register pipeline (K must be multiple of 64)
__device__ inline void gemm_mainloop(v8f (&acc)[2][4],
                                     const bf16* a0, const bf16* a1,
                                     const bf16* b0, const bf16* b1,
                                     const bf16* b2, const bf16* b3, int K) {
    TileRegs T0, T1;
    load_tile(T0, a0, a1, b0, b1, b2, b3, 0);
    for (int kk = 0; kk < K; kk += 64) {
        load_tile(T1, a0, a1, b0, b1, b2, b3, kk + 32);
        mma_tile(acc, T0);
        if (kk + 64 < K) load_tile(T0, a0, a1, b0, b1, b2, b3, kk + 64);
        mma_tile(acc, T1);
    }
}

// ============================================================================
// Kernel 1: packed QKV projection (bf16 x bf16 -> f32 acc), scatter epilogue
// into per-head layouts: Q scaled by 1/sqrt(Hd), K row-major, V transposed.
// Block = 8 waves arranged 2(M) x 4(N) -> 64(M) x 256(N) per block.
// ============================================================================
__global__ __launch_bounds__(256) void qkv_proj_kernel(
    const bf16* __restrict__ a_in, const bf16* __restrict__ w_in,
    bf16* __restrict__ qbuf, bf16* __restrict__ kbuf, bf16* __restrict__ vt) {
    const int lane = threadIdx.x & 31;
    const int wv   = threadIdx.x >> 5;
    const int col  = lane & 15;
    const int hi   = lane >> 4;
    const int m0   = blockIdx.x * 64 + (wv & 1) * 32;
    const int n0   = blockIdx.y * 256 + (wv >> 1) * 64;

    const bf16* a0 = a_in + (size_t)(m0 + col) * E_DIM + 16 * hi;
    const bf16* a1 = a_in + (size_t)(m0 + 16 + col) * E_DIM + 16 * hi;
    const bf16* b0 = w_in + (size_t)(n0 + 0 * 16 + col) * E_DIM + 16 * hi;
    const bf16* b1 = w_in + (size_t)(n0 + 1 * 16 + col) * E_DIM + 16 * hi;
    const bf16* b2 = w_in + (size_t)(n0 + 2 * 16 + col) * E_DIM + 16 * hi;
    const bf16* b3 = w_in + (size_t)(n0 + 3 * 16 + col) * E_DIM + 16 * hi;

    v8f acc[2][4] = {};
    gemm_mainloop(acc, a0, a1, b0, b1, b2, b3, E_DIM);

#pragma unroll
    for (int i = 0; i < 2; ++i) {
#pragma unroll
        for (int t = 0; t < 4; ++t) {
            const int n = n0 + t * 16 + col;
#pragma unroll
            for (int r = 0; r < 8; ++r) {
                const int m   = m0 + 16 * i + r + 8 * hi;
                const int b   = m >> 11;
                const int tok = m & (T_SEQ - 1);
                const float val = acc[i][t][r];
                if (n < E_DIM) {
                    const int h = n >> 6, d = n & 63;
                    qbuf[((size_t)(b * H_NUM + h) * T_SEQ + tok) * HD + d] = (bf16)(val * 0.125f);
                } else if (n < 2 * E_DIM) {
                    const int f = n - E_DIM, h = f >> 6, d = f & 63;
                    kbuf[((size_t)(b * H_NUM + h) * T_SEQ + tok) * HD + d] = (bf16)val;
                } else {
                    const int f = n - 2 * E_DIM, h = f >> 6, d = f & 63;
                    vt[((size_t)(b * H_NUM + h) * HD + d) * T_SEQ + tok] = (bf16)val;
                }
            }
        }
    }
}

// ============================================================================
// Kernel 2: banded flash attention with ALiBi. One wave = one (b,h,16-row
// query tile). K fragments are ping-pong double-buffered across key-tile
// iterations (prefetch issued right after the QK^T WMMAs, latency covered by
// softmax VALU + PV WMMAs); V loads issue at iteration top and hide under
// QK^T + softmax.
// ============================================================================
__global__ __launch_bounds__(256) void attn_kernel(
    const bf16* __restrict__ qbuf, const bf16* __restrict__ kbuf,
    const bf16* __restrict__ vt, bf16* __restrict__ obuf) {
    __shared__ __align__(16) bf16 pls[8 * 16 * 32];  // per-wave P transpose scratch
    const int lane = threadIdx.x & 31;
    const int wv   = threadIdx.x >> 5;
    const int wid  = blockIdx.x * 8 + wv;
    const int qt   = wid & 127;
    const int h    = (wid >> 7) & 15;
    const int b    = wid >> 11;
    const int q0   = qt * 16;
    const int col  = lane & 15;
    const int hi   = lane >> 4;
    const size_t bh = (size_t)(b * H_NUM + h);

    // Q fragments (A 16x32 per 32-wide d-chunk); Q pre-scaled by 1/sqrt(Hd)
    Frag qf[2];
    const bf16* qrow = qbuf + (bh * T_SEQ + q0 + col) * HD;
#pragma unroll
    for (int c = 0; c < 2; ++c) {
        qf[c].q[0] = *(const u32x4*)(qrow + 32 * c + 16 * hi);
        qf[c].q[1] = *(const u32x4*)(qrow + 32 * c + 16 * hi + 8);
    }

    const float slope = exp2f(-0.5f * (float)(h + 1));  // H=16 ALiBi slopes
    float mrow[8], lrow[8];
    v8f acc[4] = {};
#pragma unroll
    for (int r = 0; r < 8; ++r) { mrow[r] = -1e30f; lrow[r] = 0.0f; }

    int start = q0 - WINDOW;
    if (start < 0) start = 0;
    start &= ~31;  // 32-aligned; max jb = 2016 keeps every tile load in-bounds
    const int jend = q0 + 15;
    bf16* pw = pls + wv * (16 * 32);

    const bf16* kbase = kbuf + bh * T_SEQ * HD + 16 * hi;
    const bf16* vbase = vt + bh * HD * T_SEQ;

    // load four K B-fragments (32 keys x 64 d) for key tile at jb
    auto load_k = [&](Frag (&kf)[4], int jb) {
#pragma unroll
        for (int nt = 0; nt < 2; ++nt) {
            const bf16* krow = kbase + (size_t)(jb + nt * 16 + col) * HD;
#pragma unroll
            for (int c = 0; c < 2; ++c) {
                kf[nt * 2 + c].q[0] = *(const u32x4*)(krow + 32 * c);
                kf[nt * 2 + c].q[1] = *(const u32x4*)(krow + 32 * c + 8);
            }
        }
    };

    // one key-tile step: uses kcur, prefetches knext for jb+32 (if in range)
    auto stage = [&](const Frag (&kcur)[4], Frag (&knext)[4], int jb) {
        // V fragments for this tile (latency hides under QK^T + softmax)
        Frag vf[4];
#pragma unroll
        for (int t = 0; t < 4; ++t) {
            const bf16* vrow = vbase + (size_t)(t * 16 + col) * T_SEQ + jb + 16 * hi;
            vf[t].q[0] = *(const u32x4*)(vrow);
            vf[t].q[1] = *(const u32x4*)(vrow + 8);
        }
        // ---- S = Q K^T : two 16x16 tiles, contraction 64 split in two ----
        v8f s[2] = {};
#pragma unroll
        for (int nt = 0; nt < 2; ++nt)
#pragma unroll
            for (int c = 0; c < 2; ++c)
                s[nt] = wmma_bf16(qf[c], kcur[nt * 2 + c], s[nt]);
        // ---- prefetch next key tile's K fragments ----
        if (jb + 32 <= jend) load_k(knext, jb + 32);
        // ---- mask (causal + window) + ALiBi bias, in D-frag layout ----
        float sv[2][8];
#pragma unroll
        for (int nt = 0; nt < 2; ++nt) {
            const int j = jb + nt * 16 + col;
#pragma unroll
            for (int r = 0; r < 8; ++r) {
                const int i = q0 + r + 8 * hi;
                const bool ok = (j <= i) && ((i - j) <= WINDOW);
                sv[nt][r] = ok ? (s[nt][r] + slope * (float)j) : -1e30f;
            }
        }
        // ---- online softmax update (rows live across 16-lane groups) ----
#pragma unroll
        for (int r = 0; r < 8; ++r) {
            float tm = fmaxf(sv[0][r], sv[1][r]);
#pragma unroll
            for (int mx = 1; mx < 16; mx <<= 1) tm = fmaxf(tm, __shfl_xor(tm, mx, 32));
            const float mnew  = fmaxf(mrow[r], tm);
            const float alpha = __expf(mrow[r] - mnew);
            const float p0 = (sv[0][r] > -1e29f) ? __expf(sv[0][r] - mnew) : 0.0f;
            const float p1 = (sv[1][r] > -1e29f) ? __expf(sv[1][r] - mnew) : 0.0f;
            float ps = p0 + p1;
#pragma unroll
            for (int mx = 1; mx < 16; mx <<= 1) ps += __shfl_xor(ps, mx, 32);
            lrow[r] = lrow[r] * alpha + ps;
            mrow[r] = mnew;
#pragma unroll
            for (int t = 0; t < 4; ++t) acc[t][r] *= alpha;
            const int row = r + 8 * hi;
            pw[row * 32 + col]      = (bf16)p0;
            pw[row * 32 + 16 + col] = (bf16)p1;
        }
        // ---- P as A-fragment (DS ops are in-order within a wave) ----
        Frag pf;
        pf.q[0] = *(const u32x4*)(pw + col * 32 + 16 * hi);
        pf.q[1] = *(const u32x4*)(pw + col * 32 + 16 * hi + 8);
        // ---- O += P V ----
#pragma unroll
        for (int t = 0; t < 4; ++t) acc[t] = wmma_bf16(pf, vf[t], acc[t]);
    };

    // ping-pong over key tiles
    Frag K0[4], K1[4];
    int jb = start;
    load_k(K0, jb);
    while (true) {
        stage(K0, K1, jb);
        jb += 32;
        if (jb > jend) break;
        stage(K1, K0, jb);
        jb += 32;
        if (jb > jend) break;
    }

    // ---- epilogue: normalize, store (B*T, E) bf16 ----
#pragma unroll
    for (int t = 0; t < 4; ++t) {
        const int cc = h * 64 + t * 16 + col;
#pragma unroll
        for (int r = 0; r < 8; ++r) {
            const int i = q0 + r + 8 * hi;
            obuf[((size_t)b * T_SEQ + i) * E_DIM + cc] = (bf16)(acc[t][r] / lrow[r]);
        }
    }
}

// ============================================================================
// Kernel 3: output projection (bf16 x bf16 -> f32 out), same GEMM mainloop.
// ============================================================================
__global__ __launch_bounds__(256) void out_proj_kernel(
    const bf16* __restrict__ a_in, const bf16* __restrict__ w,
    float* __restrict__ out) {
    const int lane = threadIdx.x & 31;
    const int wv   = threadIdx.x >> 5;
    const int col  = lane & 15;
    const int hi   = lane >> 4;
    const int m0   = blockIdx.x * 64 + (wv & 1) * 32;
    const int n0   = blockIdx.y * 256 + (wv >> 1) * 64;

    const bf16* a0 = a_in + (size_t)(m0 + col) * E_DIM + 16 * hi;
    const bf16* a1 = a_in + (size_t)(m0 + 16 + col) * E_DIM + 16 * hi;
    const bf16* b0 = w + (size_t)(n0 + 0 * 16 + col) * E_DIM + 16 * hi;
    const bf16* b1 = w + (size_t)(n0 + 1 * 16 + col) * E_DIM + 16 * hi;
    const bf16* b2 = w + (size_t)(n0 + 2 * 16 + col) * E_DIM + 16 * hi;
    const bf16* b3 = w + (size_t)(n0 + 3 * 16 + col) * E_DIM + 16 * hi;

    v8f acc[2][4] = {};
    gemm_mainloop(acc, a0, a1, b0, b1, b2, b3, E_DIM);

#pragma unroll
    for (int i = 0; i < 2; ++i)
#pragma unroll
        for (int t = 0; t < 4; ++t) {
            const int n = n0 + t * 16 + col;
#pragma unroll
            for (int r = 0; r < 8; ++r)
                out[(size_t)(m0 + 16 * i + r + 8 * hi) * E_DIM + n] = acc[i][t][r];
        }
}

// ============================================================================
extern "C" void kernel_launch(void* const* d_in, const int* in_sizes, int n_in,
                              void* d_out, int out_size, void* d_ws, size_t ws_size,
                              hipStream_t stream) {
    const float* query = (const float*)d_in[0];   // (B,T,E)  f32
    const float* w_in  = (const float*)d_in[1];   // (3E,E)   f32
    const float* w_out = (const float*)d_in[2];   // (E,E)    f32
    float* out = (float*)d_out;                   // (B,T,E)  f32

    // workspace (bf16): query | Win | Wout | Q | K | V^T | attn-out
    char* ws = (char*)d_ws;
    bf16* qxbf = (bf16*)(ws);                     // 8 MB  (4096x1024)
    bf16* wibf = (bf16*)(ws + (8ull  << 20));     // 6 MB  (3072x1024)
    bf16* wobf = (bf16*)(ws + (14ull << 20));     // 2 MB  (1024x1024)
    bf16* qbuf = (bf16*)(ws + (16ull << 20));     // 8 MB
    bf16* kbuf = (bf16*)(ws + (24ull << 20));     // 8 MB
    bf16* vt   = (bf16*)(ws + (32ull << 20));     // 8 MB
    bf16* obuf = (bf16*)(ws + (40ull << 20));     // 8 MB

    // one-time f32 -> bf16 pre-conversion (weights + activations)
    cvt_bf16_kernel<<<dim3(1024), 256, 0, stream>>>(query, qxbf, (B_NUM * T_SEQ * E_DIM) / 4);
    cvt_bf16_kernel<<<dim3(1024), 256, 0, stream>>>(w_in,  wibf, (3 * E_DIM * E_DIM) / 4);
    cvt_bf16_kernel<<<dim3(512),  256, 0, stream>>>(w_out, wobf, (E_DIM * E_DIM) / 4);

    // QKV: M=4096, N=3072 -> grid (4096/64, 3072/256)
    qkv_proj_kernel<<<dim3(64, 12), 256, 0, stream>>>(qxbf, wibf, qbuf, kbuf, vt);
    // Attention: B*H*(T/16) = 4096 waves / 8 per block
    attn_kernel<<<dim3(512), 256, 0, stream>>>(qbuf, kbuf, vt, obuf);
    // Out proj: M=4096, N=1024 -> grid (4096/64, 1024/256)
    out_proj_kernel<<<dim3(64, 4), 256, 0, stream>>>(obuf, wobf, out);
}
